// FlashMoEModel_61916248539797
// MI455X (gfx1250) — matrix operand order repaired and checked by reading, not compiled, not run on hardware
//
#include <hip/hip_runtime.h>
#include <hip/hip_bf16.h>
#include <math.h>

#define D_MODEL 1024
#define NEXP    16
#define RANK    64
#define BATCH   32768
#define CAP     8192   // per-expert token capacity (mean 4096, ~68 sigma headroom)

typedef __bf16 bf16_t;
typedef __attribute__((ext_vector_type(16))) __bf16    v16bf;
typedef __attribute__((ext_vector_type(8)))  __bf16    v8bf;
typedef __attribute__((ext_vector_type(8)))  float     v8f;
typedef __attribute__((ext_vector_type(4)))  unsigned  v4u;
typedef __attribute__((ext_vector_type(8)))  int       v8i;
typedef __attribute__((ext_vector_type(4)))  int       v4i;

// ---------------- WMMA helpers (CDNA5 16x16x32 bf16, f32 accum) ----------------

__device__ __forceinline__ v8f wmma_bf16(v16bf a, v16bf b, v8f c) {
  return __builtin_amdgcn_wmma_f32_16x16x32_bf16(false, a, false, b, (short)0, c,
                                                 false, false);
}

__device__ __forceinline__ v16bf cat8(v8bf lo, v8bf hi) {
  v16bf r;
#pragma unroll
  for (int i = 0; i < 8; ++i) { r[i] = lo[i]; r[i + 8] = hi[i]; }
  return r;
}

// B fragment: 16 contiguous bf16 at p (lane holds n=lane%16, K=16*half+{0..15})
__device__ __forceinline__ v16bf load_b_frag(const bf16_t* p) {
  const v8bf* q = (const v8bf*)p;
  return cat8(q[0], q[1]);
}

// A fragment: 16 contiguous bf16 at p (K=16*half+{0..15})
__device__ __forceinline__ v16bf load_a_frag(const bf16_t* p) {
  return cat8(*(const v8bf*)p, *(const v8bf*)(p + 8));
}

// A fragment straight from fp32 memory (inline convert)
__device__ __forceinline__ v16bf load_a_f32(const float* p) {
  v16bf r;
#pragma unroll
  for (int i = 0; i < 8; ++i) {
    r[i]     = (__bf16)p[i];
    r[i + 8] = (__bf16)p[8 + i];
  }
  return r;
}

// ---------------- TDM: 2D tile  global -> LDS  (bf16 elements) ----------------
// Tensor: rows of length tensor_d0 elems (contiguous), row stride stride0 elems.
// Tile: tile_d0 elems x tile_d1 rows, starting at gaddr; rows land back-to-back
// in LDS starting at lds_byte_addr (row y -> lds + y*tile_d0*2 bytes).

__device__ __forceinline__ void tdm_load_2d(unsigned lds_byte_addr,
                                            const bf16_t* gaddr,
                                            unsigned tensor_d0, unsigned tensor_d1,
                                            unsigned stride0,
                                            unsigned tile_d0, unsigned tile_d1) {
  unsigned long long ga = (unsigned long long)(uintptr_t)gaddr;
  v4u g0;
  g0[0] = 1u;                                            // count=1, user descriptor
  g0[1] = lds_byte_addr;                                 // LDS dest (bytes)
  g0[2] = (unsigned)(ga & 0xFFFFFFFFu);                  // global addr lo
  g0[3] = (unsigned)((ga >> 32) & 0x01FFFFFFu) | (2u << 30);  // addr hi | type=2
  v8i g1;
  g1[0] = (int)(1u << 16);                               // data_size=1 (2 bytes)
  g1[1] = (int)((tensor_d0 & 0xFFFFu) << 16);            // tensor_dim0[15:0]
  g1[2] = (int)(((tensor_d0 >> 16) & 0xFFFFu) |
                ((tensor_d1 & 0xFFFFu) << 16));          // dim0 hi | dim1 lo
  g1[3] = (int)(((tensor_d1 >> 16) & 0xFFFFu) |
                ((tile_d0 & 0xFFFFu) << 16));            // dim1 hi | tile_dim0
  g1[4] = (int)(tile_d1 & 0xFFFFu);                      // tile_dim1 (tile_dim2=0)
  g1[5] = (int)stride0;                                  // tensor_dim0_stride lo
  g1[6] = 0;                                             // stride0 hi | stride1 lo
  g1[7] = 0;
  v4i z4 = {0, 0, 0, 0};
#if defined(__clang_major__) && (__clang_major__ >= 23)
  v8i z8 = {0, 0, 0, 0, 0, 0, 0, 0};
  __builtin_amdgcn_tensor_load_to_lds(g0, g1, z4, z4, z8, 0);
#else
  __builtin_amdgcn_tensor_load_to_lds(g0, g1, z4, z4, 0);
#endif
}

__device__ __forceinline__ unsigned lds_addr_of(const void* p) {
  return (unsigned)(uintptr_t)p;   // low 32 bits of flat shared address = LDS offset
}

// ---------------- K1: fp32 -> bf16 conversion / counter zero ----------------

__global__ void k_f2bf(const float* __restrict__ in, bf16_t* __restrict__ out, int n) {
  for (int i = blockIdx.x * 256 + threadIdx.x; i < n; i += gridDim.x * 256)
    out[i] = (__bf16)in[i];
}

__global__ void k_zero_counts(int* __restrict__ c) {
  if (threadIdx.x < NEXP) c[threadIdx.x] = 0;
}

// ---------------- K2: encoder GEMM  enc = x @ We^T + be ----------------
// WG = 8 waves (2 row x 4 col). Wave tile 32x64, block tile 64x256.
// B (We) staged to LDS by TDM in 64-K double-buffered slices; A converted inline.

__global__ __launch_bounds__(256) void k_encoder(
    const float* __restrict__ x, const bf16_t* __restrict__ We,
    const float* __restrict__ be, float* __restrict__ out,
    bf16_t* __restrict__ enc_bf) {
  __shared__ __align__(16) bf16_t bt[2][256 * 64];   // 2 x 32KB: [col][k64]
  const int lane = threadIdx.x & 31;
  const int wave = threadIdx.x >> 5;
  const int wr = wave >> 2, wc = wave & 3;
  const int row0 = blockIdx.x * 64 + wr * 32;
  const int col0 = blockIdx.y * 256;
  const int m = lane & 15, hi = lane >> 4;

  v8f acc[2][4];
#pragma unroll
  for (int a = 0; a < 2; ++a)
#pragma unroll
    for (int b = 0; b < 4; ++b) acc[a][b] = (v8f)0.0f;

  const float* xA0 = x + (size_t)(row0 + m) * D_MODEL + 16 * hi;
  const float* xA1 = xA0 + (size_t)16 * D_MODEL;

  const int NSTAGE = D_MODEL / 64;                   // 16
  if (wave == 0)
    tdm_load_2d(lds_addr_of(&bt[0][0]), We + (size_t)col0 * D_MODEL,
                D_MODEL, 256, D_MODEL, 64, 256);

  for (int s = 0; s < NSTAGE; ++s) {
    if (wave == 0) {
      if (s + 1 < NSTAGE) {
        tdm_load_2d(lds_addr_of(&bt[(s + 1) & 1][0]),
                    We + (size_t)col0 * D_MODEL + (s + 1) * 64,
                    D_MODEL, 256, D_MODEL, 64, 256);
        __builtin_amdgcn_s_wait_tensorcnt(1);        // current slice complete
      } else {
        __builtin_amdgcn_s_wait_tensorcnt(0);        // last slice complete
      }
    }
    __syncthreads();                                 // staged slice visible

    const bf16_t* bb = &bt[s & 1][0];
    const int kb = s * 64;
#pragma unroll
    for (int kk = 0; kk < 64; kk += 32) {
      v16bf a0 = load_a_f32(xA0 + kb + kk);
      v16bf a1 = load_a_f32(xA1 + kb + kk);
#pragma unroll
      for (int ni = 0; ni < 4; ++ni) {
        // LDS row (local col) stride = 64 elems; frag K offset kk + 16*hi
        v16bf bfr = load_b_frag(bb + (size_t)(wc * 64 + ni * 16 + m) * 64 + kk + 16 * hi);
        acc[0][ni] = wmma_bf16(a0, bfr, acc[0][ni]);
        acc[1][ni] = wmma_bf16(a1, bfr, acc[1][ni]);
      }
    }
    __syncthreads();                                 // done reading before overwrite
  }

  // C/D layout: lane n = col0+wc*64+ni*16+m ; vgpr i holds row (mi*16 + hi*8 + i)
#pragma unroll
  for (int ni = 0; ni < 4; ++ni) {
    const int n = col0 + wc * 64 + ni * 16 + m;
    const float bias = be[n];
#pragma unroll
    for (int mi = 0; mi < 2; ++mi) {
#pragma unroll
      for (int i = 0; i < 8; ++i) {
        const int r = row0 + mi * 16 + hi * 8 + i;
        const float v = acc[mi][ni][i] + bias;
        out[(size_t)r * D_MODEL + n] = v;
        enc_bf[(size_t)r * D_MODEL + n] = (__bf16)v;
      }
    }
  }
}

// ---------------- K3: gating (logits, top-2, softmax, expert lists) ----------------

__global__ __launch_bounds__(256) void k_gate(
    const bf16_t* __restrict__ enc, const bf16_t* __restrict__ Wg,
    const float* __restrict__ gamma, int* __restrict__ counts,
    int* __restrict__ tok_idx, float* __restrict__ tok_w) {
  const int lane = threadIdx.x & 31;
  const int wave = threadIdx.x >> 5;
  const int b = blockIdx.x * 8 + wave;

  float acc[NEXP];
#pragma unroll
  for (int e = 0; e < NEXP; ++e) acc[e] = 0.0f;

  const bf16_t* er = enc + (size_t)b * D_MODEL;
  for (int d = lane; d < D_MODEL; d += 32) {
    const float xv = (float)er[d];
#pragma unroll
    for (int e = 0; e < NEXP; ++e)
      acc[e] += xv * (float)Wg[e * D_MODEL + d];
  }
#pragma unroll
  for (int e = 0; e < NEXP; ++e) {
#pragma unroll
    for (int off = 16; off > 0; off >>= 1)
      acc[e] += __shfl_xor(acc[e], off, 32);
  }

  if (lane == 0) {
    int e1 = 0; float l1 = acc[0];
#pragma unroll
    for (int e = 1; e < NEXP; ++e)
      if (acc[e] > l1) { l1 = acc[e]; e1 = e; }   // strict > keeps lowest index on ties
    int e2 = -1; float l2 = -3.4e38f;
#pragma unroll
    for (int e = 0; e < NEXP; ++e)
      if (e != e1 && acc[e] > l2) { l2 = acc[e]; e2 = e; }

    const float ex = __expf(l2 - l1);             // TAU = 1
    const float denom = 1.0f + ex + 1e-12f;
    const float w1 = 1.0f / denom;
    float w2 = ex / denom;
    if (w2 <= 1e-12f) w2 = 0.0f;                  // keep_mask

    const int p1 = atomicAdd(&counts[e1], 1);
    if (p1 < CAP) {
      tok_idx[e1 * CAP + p1] = b;
      tok_w[e1 * CAP + p1] = w1 * gamma[e1];
    }
    if (w2 > 0.0f) {
      const int p2 = atomicAdd(&counts[e2], 1);
      if (p2 < CAP) {
        tok_idx[e2 * CAP + p2] = b;
        tok_w[e2 * CAP + p2] = w2 * gamma[e2];
      }
    }
  }
}

// ---------------- K4: hidden = w*gamma * silu(enc[tok] @ U_e^T) ----------------
// Whole U_e (64x1024 bf16 = 128KB) TDM-loaded once into LDS; all 8 waves share it.

__global__ __launch_bounds__(256) void k_hidden(
    const bf16_t* __restrict__ enc, const bf16_t* __restrict__ U,
    const int* __restrict__ counts, const int* __restrict__ tok_idx,
    const float* __restrict__ tok_w, bf16_t* __restrict__ hidden) {
  __shared__ __align__(16) bf16_t ut[RANK * D_MODEL];  // 128KB, identity copy of U_e
  const int e = blockIdx.y;
  int cnt = counts[e]; if (cnt > CAP) cnt = CAP;
  if ((int)blockIdx.x * 128 >= cnt) return;            // block-uniform early out

  const int lane = threadIdx.x & 31;
  const int wave = threadIdx.x >> 5;
  if (wave == 0) {
    tdm_load_2d(lds_addr_of(&ut[0]), U + (size_t)e * (RANK * D_MODEL),
                D_MODEL, RANK, D_MODEL, D_MODEL, RANK);
    __builtin_amdgcn_s_wait_tensorcnt(0);
  }
  __syncthreads();

  const int t0 = blockIdx.x * 128 + wave * 16;
  if (t0 >= cnt) return;                               // wave-uniform, after barrier

  const int m = lane & 15, hi = lane >> 4;
  const int t = t0 + m;
  const int tok = (t < cnt) ? tok_idx[e * CAP + t] : 0;
  const bf16_t* arow = enc + (size_t)tok * D_MODEL + 16 * hi;

  v8f acc[4];
#pragma unroll
  for (int ni = 0; ni < 4; ++ni) acc[ni] = (v8f)0.0f;

  for (int kb = 0; kb < D_MODEL; kb += 32) {
    v16bf a = load_a_frag(arow + kb);
#pragma unroll
    for (int ni = 0; ni < 4; ++ni) {
      v16bf bfr = load_b_frag(ut + (size_t)(ni * 16 + m) * D_MODEL + kb + 16 * hi);
      acc[ni] = wmma_bf16(a, bfr, acc[ni]);
    }
  }

#pragma unroll
  for (int ni = 0; ni < 4; ++ni) {
    const int n = ni * 16 + m;
#pragma unroll
    for (int i = 0; i < 8; ++i) {
      const int tt = t0 + hi * 8 + i;
      if (tt < cnt) {
        const float v = acc[ni][i];
        const float s = v / (1.0f + __expf(-v));       // silu
        const float w = tok_w[e * CAP + tt];
        hidden[((size_t)e * CAP + tt) * RANK + n] = (__bf16)(s * w);
      }
    }
  }
}

// ---------------- K5: out[tok] += hidden @ V_e^T   (K = 64) ----------------
// Whole V_e (1024x64 bf16 = 128KB) TDM-loaded once into LDS; wave keeps A in regs.

__global__ __launch_bounds__(256) void k_delta(
    const bf16_t* __restrict__ hidden, const bf16_t* __restrict__ V,
    const int* __restrict__ counts, const int* __restrict__ tok_idx,
    float* __restrict__ out) {
  __shared__ __align__(16) bf16_t vt[D_MODEL * RANK];  // 128KB, identity copy of V_e
  const int e = blockIdx.y;
  int cnt = counts[e]; if (cnt > CAP) cnt = CAP;
  if ((int)blockIdx.x * 128 >= cnt) return;            // block-uniform early out

  const int lane = threadIdx.x & 31;
  const int wave = threadIdx.x >> 5;
  if (wave == 0) {
    tdm_load_2d(lds_addr_of(&vt[0]), V + (size_t)e * (D_MODEL * RANK),
                RANK, D_MODEL, RANK, RANK, D_MODEL);
    __builtin_amdgcn_s_wait_tensorcnt(0);
  }
  __syncthreads();

  const int t0 = blockIdx.x * 128 + wave * 16;
  if (t0 >= cnt) return;                               // wave-uniform, after barrier

  const int m = lane & 15, hi = lane >> 4;
  const bf16_t* hrow = hidden + ((size_t)e * CAP + t0 + m) * RANK + 16 * hi;
  v16bf a0 = load_a_frag(hrow);                        // K 0..31
  v16bf a1 = load_a_frag(hrow + 32);                   // K 32..63

  int tk[8];
#pragma unroll
  for (int i = 0; i < 8; ++i) {
    const int tt = t0 + hi * 8 + i;
    tk[i] = (tt < cnt) ? tok_idx[e * CAP + tt] : -1;
  }

#pragma unroll 2
  for (int nf = 0; nf < D_MODEL / 16; ++nf) {
    const int n = nf * 16 + m;
    v16bf b0 = load_b_frag(vt + (size_t)n * RANK + 16 * hi);       // K 16*hi+{0..15}
    v16bf b1 = load_b_frag(vt + (size_t)n * RANK + 32 + 16 * hi);  // K 32+...
    v8f acc = (v8f)0.0f;
    acc = wmma_bf16(a0, b0, acc);
    acc = wmma_bf16(a1, b1, acc);
#pragma unroll
    for (int i = 0; i < 8; ++i)
      if (tk[i] >= 0)
        unsafeAtomicAdd(&out[(size_t)tk[i] * D_MODEL + n], acc[i]);
  }
}

// ---------------- host launcher ----------------

extern "C" void kernel_launch(void* const* d_in, const int* in_sizes, int n_in,
                              void* d_out, int out_size, void* d_ws, size_t ws_size,
                              hipStream_t stream) {
  const float* x     = (const float*)d_in[0];
  const float* We    = (const float*)d_in[1];
  const float* be    = (const float*)d_in[2];
  const float* Wg    = (const float*)d_in[3];
  const float* U     = (const float*)d_in[4];
  const float* V     = (const float*)d_in[5];
  const float* gamma = (const float*)d_in[6];
  float* out = (float*)d_out;

  // workspace carve-up (~87 MB)
  char* w = (char*)d_ws;
  bf16_t* We_bf  = (bf16_t*)w; w += (size_t)D_MODEL * D_MODEL * 2;
  bf16_t* Wg_bf  = (bf16_t*)w; w += (size_t)NEXP * D_MODEL * 2;
  bf16_t* U_bf   = (bf16_t*)w; w += (size_t)NEXP * RANK * D_MODEL * 2;
  bf16_t* V_bf   = (bf16_t*)w; w += (size_t)NEXP * D_MODEL * RANK * 2;
  bf16_t* enc_bf = (bf16_t*)w; w += (size_t)BATCH * D_MODEL * 2;
  bf16_t* hidden = (bf16_t*)w; w += (size_t)NEXP * CAP * RANK * 2;
  int* counts    = (int*)w;    w += 64;
  int* tok_idx   = (int*)w;    w += (size_t)NEXP * CAP * 4;
  float* tok_w   = (float*)w;

  k_f2bf<<<512, 256, 0, stream>>>(We, We_bf, D_MODEL * D_MODEL);
  k_f2bf<<<64, 256, 0, stream>>>(Wg, Wg_bf, NEXP * D_MODEL);
  k_f2bf<<<512, 256, 0, stream>>>(U, U_bf, NEXP * RANK * D_MODEL);
  k_f2bf<<<512, 256, 0, stream>>>(V, V_bf, NEXP * D_MODEL * RANK);
  k_zero_counts<<<1, 32, 0, stream>>>(counts);

  dim3 g2(BATCH / 64, D_MODEL / 256);
  k_encoder<<<g2, 256, 0, stream>>>(x, We_bf, be, out, enc_bf);

  k_gate<<<BATCH / 8, 256, 0, stream>>>(enc_bf, Wg_bf, gamma, counts, tok_idx, tok_w);

  dim3 g4(CAP / 128, NEXP);
  k_hidden<<<g4, 256, 0, stream>>>(enc_bf, U_bf, counts, tok_idx, tok_w, hidden);
  k_delta<<<g4, 256, 0, stream>>>(hidden, V_bf, counts, tok_idx, out);
}
